// JasMultipleBody_50156628083360
// MI455X (gfx1250) — matrix operand order repaired and checked by reading, not compiled, not run on hardware
//
#include <hip/hip_runtime.h>

// ---------------------------------------------------------------------------
// Problem constants (reference: NS=16384 samples, N=2048 sites, n=4)
// ---------------------------------------------------------------------------
#define NSAMP 16384
#define NSITE 2048
#define BM 128
#define BN 128
#define BK 32
#define NK (NSITE / BK)   // 64 K-steps (even, required by the 2x unroll)
#define LSTR 40           // LDS row stride in bf16 elements (32 + 8 pad)
#define NTHREADS 256
#define WTILE (BN * LSTR) // one W LDS buffer in ushorts

typedef __attribute__((ext_vector_type(16))) __bf16 v16bf;
typedef __attribute__((ext_vector_type(8)))  float  v8f;

struct alignas(8) us4 { unsigned short x, y, z, w; };

union Frag { v16bf v; uint4 q[2]; };

// Round-to-nearest-even f32 -> bf16 (bit pattern)
__device__ __forceinline__ unsigned short bf16_rne(float f) {
    unsigned u = __float_as_uint(f);
    u += 0x7FFFu + ((u >> 16) & 1u);
    return (unsigned short)(u >> 16);
}

// Split f32 into bf16 hi + bf16 lo residual (hi + lo ~= f, ~16 mantissa bits)
__device__ __forceinline__ void bf16_split(float f, unsigned short& hs, unsigned short& ls) {
    hs = bf16_rne(f);
    float hf = __uint_as_float((unsigned)hs << 16);
    ls = bf16_rne(f - hf);
}

// CDNA5 async global->LDS copy (ASYNCcnt-tracked, bypasses VGPRs).
// Generic shared pointer low 32 bits == LDS byte offset (aperture rule).
__device__ __forceinline__ void async_copy_b128(unsigned short* lds_dst,
                                                const unsigned short* gsrc) {
    unsigned lds_off = (unsigned)(size_t)lds_dst;
    asm volatile("global_load_async_to_lds_b128 %0, %1, off"
                 :: "v"(lds_off), "v"(gsrc)
                 : "memory");
}

// Workgroup barrier draining ONLY DScnt + ASYNCcnt (<= MAXA). Unlike
// __syncthreads(), this does NOT wait LOADcnt to 0, so the prefetched
// global A-loads for the next K-step stay in flight through compute.
template <int MAXA>
__device__ __forceinline__ void barrier_fill() {
    asm volatile(
        "s_wait_dscnt 0x0\n\t"
        "s_wait_asynccnt %0\n\t"
        "s_barrier_signal -1\n\t"
        "s_barrier_wait -1"
        :: "i"(MAXA) : "memory");
}

__device__ __forceinline__ void barrier_drain() {
    asm volatile(
        "s_wait_dscnt 0x0\n\t"
        "s_barrier_signal -1\n\t"
        "s_barrier_wait -1"
        ::: "memory");
}

// ---------------------------------------------------------------------------
// Expand packed strictly-lower-triangular kernel vector into dense W,
// stored as bf16 hi/lo pair. W[j][i] = kernel[j*(j-1)/2 + i] for i<j else 0.
// ---------------------------------------------------------------------------
__global__ __launch_bounds__(NTHREADS) void build_w(
    const float* __restrict__ kern,
    unsigned short* __restrict__ Whi,
    unsigned short* __restrict__ Wlo)
{
    size_t idx = (size_t)blockIdx.x * NTHREADS + threadIdx.x;
    int j = (int)(idx / NSITE);
    int i = (int)(idx % NSITE);
    float v = 0.0f;
    if (i < j) v = kern[((size_t)j * (j - 1)) / 2 + i];
    unsigned short hs, ls;
    bf16_split(v, hs, ls);
    Whi[idx] = hs;
    Wlo[idx] = ls;
}

__global__ __launch_bounds__(NTHREADS) void zero_out(float* __restrict__ out) {
    out[(size_t)blockIdx.x * NTHREADS + threadIdx.x] = 0.0f;
}

// ---------------------------------------------------------------------------
// One GEMM stage: C = A @ W^T with A = x (USE_Z=0) or A = zin*x.
// C[m][n] = sum_k A[m][k] * W[n][k] -> B tile is row-major W (contiguous k),
// async-copied to double-buffered LDS. Split precision: Ah*Bh+Ah*Bl+Al*Bh.
// FUSE_DOT=1: accumulate out[row] += sum_n C[m][n]*x[m][n] instead of storing C.
// ---------------------------------------------------------------------------
template <int USE_Z, int FUSE_DOT>
__global__ __launch_bounds__(NTHREADS) void gemm_stage(
    const float* __restrict__ x,
    const float* __restrict__ zin,
    const unsigned short* __restrict__ Whi,
    const unsigned short* __restrict__ Wlo,
    float* __restrict__ zout,
    float* __restrict__ out)
{
    __shared__ unsigned short sAh[BM * LSTR];
    __shared__ unsigned short sAl[BM * LSTR];
    __shared__ unsigned short sBh[2 * WTILE];   // double-buffered async W tiles
    __shared__ unsigned short sBl[2 * WTILE];

    // N-tiles on x (fast axis) so one M-slab of A is reused across all its
    // column blocks out of L2; W (16MB) is L2-resident throughout.
    const int n0 = blockIdx.x * BN;
    const int m0 = blockIdx.y * BM;

    const int tid  = threadIdx.x;
    const int lane = tid & 31;
    const int wid  = tid >> 5;      // 8 waves
    const int wm   = wid >> 2;      // 0..1  (64 rows each)
    const int wn   = wid & 3;       // 0..3  (32 cols each)
    const int l15  = lane & 15;
    const int lh   = lane >> 4;

    v8f acc[4][2] = {};

    // Per-thread A-tile mapping: 4 x float4 covering 128x32 f32.
    int ar[4], ak[4];
#pragma unroll
    for (int it = 0; it < 4; ++it) {
        int li = tid + it * NTHREADS;       // 0..1023
        ar[it] = li >> 3;                   // row 0..127
        ak[it] = (li & 7) << 2;             // k 0..28
    }
    // Per-thread W-tile mapping: 2 x b128 per matrix covering 128x32 bf16.
    int wr[2], wk[2];
#pragma unroll
    for (int it = 0; it < 2; ++it) {
        int li = tid + it * NTHREADS;       // 0..511
        wr[it] = li >> 2;                   // row 0..127
        wk[it] = (li & 3) << 3;             // k 0,8,16,24
    }

    // Two disjoint prefetch register sets (ping-pong) so in-flight global
    // loads never collide with compute fragment registers.
    float4 axA[4], azA[4], axB[4], azB[4];

    auto globA = [&](int k0, float4 (&axp)[4], float4 (&azp)[4]) {
#pragma unroll
        for (int it = 0; it < 4; ++it) {
            axp[it] = *(const float4*)(x + (size_t)(m0 + ar[it]) * NSITE + k0 + ak[it]);
            if (USE_Z)
                azp[it] = *(const float4*)(zin + (size_t)(m0 + ar[it]) * NSITE + k0 + ak[it]);
        }
    };

    auto storeA = [&](float4 (&axp)[4], float4 (&azp)[4]) {
#pragma unroll
        for (int it = 0; it < 4; ++it) {
            float4 av = axp[it];
            if (USE_Z) {
                av.x *= azp[it].x; av.y *= azp[it].y; av.z *= azp[it].z; av.w *= azp[it].w;
            }
            us4 h, l;
            bf16_split(av.x, h.x, l.x);
            bf16_split(av.y, h.y, l.y);
            bf16_split(av.z, h.z, l.z);
            bf16_split(av.w, h.w, l.w);
            *(us4*)&sAh[ar[it] * LSTR + ak[it]] = h;
            *(us4*)&sAl[ar[it] * LSTR + ak[it]] = l;
        }
    };

    auto asyncW = [&](int k0, int buf) {
#pragma unroll
        for (int it = 0; it < 2; ++it) {
            size_t goff = (size_t)(n0 + wr[it]) * NSITE + k0 + wk[it];
            int doff = buf * WTILE + wr[it] * LSTR + wk[it];
            async_copy_b128(&sBh[doff], Whi + goff);
            async_copy_b128(&sBl[doff], Wlo + goff);
        }
    };

    auto compute = [&](int buf) {
        Frag bh[2], bl[2];
#pragma unroll
        for (int fn = 0; fn < 2; ++fn) {
            int rr = buf * WTILE + (wn * 32 + fn * 16 + l15) * LSTR + lh * 16;
            const uint4* ph = (const uint4*)&sBh[rr];
            bh[fn].q[0] = ph[0]; bh[fn].q[1] = ph[1];
            const uint4* pl = (const uint4*)&sBl[rr];
            bl[fn].q[0] = pl[0]; bl[fn].q[1] = pl[1];
        }
#pragma unroll
        for (int fm = 0; fm < 4; ++fm) {
            int ra = wm * 64 + fm * 16 + l15;
            int ca = lh * 16;
            Frag ah, al;
            ah.q[0] = *(const uint4*)&sAh[ra * LSTR + ca];
            ah.q[1] = *(const uint4*)&sAh[ra * LSTR + ca + 8];
            al.q[0] = *(const uint4*)&sAl[ra * LSTR + ca];
            al.q[1] = *(const uint4*)&sAl[ra * LSTR + ca + 8];
#pragma unroll
            for (int fn = 0; fn < 2; ++fn) {
                acc[fm][fn] = __builtin_amdgcn_wmma_f32_16x16x32_bf16(
                    false, ah.v, false, bh[fn].v, (short)0, acc[fm][fn], false, false);
                acc[fm][fn] = __builtin_amdgcn_wmma_f32_16x16x32_bf16(
                    false, ah.v, false, bl[fn].v, (short)0, acc[fm][fn], false, false);
                acc[fm][fn] = __builtin_amdgcn_wmma_f32_16x16x32_bf16(
                    false, al.v, false, bh[fn].v, (short)0, acc[fm][fn], false, false);
            }
        }
    };

    // --- pipelined K loop, unrolled by 2 with ping-pong A register sets:
    //   A(k+1): global->VGPR issued before compute(k), consumed by storeA(k+1).
    //   W(k+1): async-to-LDS issued at top of sub-iter k into the other LDS
    //           buffer, landing while compute(k) runs; fill waits asynccnt<=4.
    globA(0, axA, azA);
    asyncW(0, 0);
    for (int kt = 0; kt < NK; kt += 2) {
        // even sub-iteration: W buffer 0, A register set A
        asyncW(kt + 1, 1);
        storeA(axA, azA);
        globA((kt + 1) * BK, axB, azB);
        barrier_fill<4>();
        compute(0);
        barrier_drain();

        // odd sub-iteration: W buffer 1, A register set B
        if (kt + 2 < NK) {
            asyncW(kt + 2, 0);
            storeA(axB, azB);
            globA((kt + 2) * BK, axA, azA);
            barrier_fill<4>();
        } else {
            storeA(axB, azB);
            barrier_fill<0>();
        }
        compute(1);
        barrier_drain();
    }

    if (FUSE_DOT) {
        // out[row] += sum over this block's columns of C[row][col]*x[row][col]
#pragma unroll
        for (int fm = 0; fm < 4; ++fm) {
#pragma unroll
            for (int v = 0; v < 8; ++v) {
                int row = m0 + wm * 64 + fm * 16 + lh * 8 + v;
                float s = 0.0f;
#pragma unroll
                for (int fn = 0; fn < 2; ++fn) {
                    int col = n0 + wn * 32 + fn * 16 + l15;
                    s += acc[fm][fn][v] * x[(size_t)row * NSITE + col];
                }
                s += __shfl_xor(s, 1, 32);
                s += __shfl_xor(s, 2, 32);
                s += __shfl_xor(s, 4, 32);
                s += __shfl_xor(s, 8, 32);
                if (l15 == 0) atomicAdd(&out[row], s);
            }
        }
    } else {
#pragma unroll
        for (int fm = 0; fm < 4; ++fm)
#pragma unroll
            for (int fn = 0; fn < 2; ++fn)
#pragma unroll
                for (int v = 0; v < 8; ++v) {
                    int row = m0 + wm * 64 + fm * 16 + lh * 8 + v;
                    int col = n0 + wn * 32 + fn * 16 + l15;
                    zout[(size_t)row * NSITE + col] = acc[fm][fn][v];
                }
    }
}

// ---------------------------------------------------------------------------
// Workspace layout:
//   [0, 8MB)      W_hi (bf16)   [8, 16MB)  W_lo (bf16)
//   [16, 144MB)   z0 (f32)      [144, 272MB) z1 (f32)
// n fixed at 4 -> 3 GEMM stages; stage 3 fuses the final row-dot via atomics.
// ---------------------------------------------------------------------------
extern "C" void kernel_launch(void* const* d_in, const int* in_sizes, int n_in,
                              void* d_out, int out_size, void* d_ws, size_t ws_size,
                              hipStream_t stream) {
    const float* x    = (const float*)d_in[0];
    const float* kern = (const float*)d_in[1];
    float*       out  = (float*)d_out;

    char* ws = (char*)d_ws;
    unsigned short* Whi = (unsigned short*)ws;
    unsigned short* Wlo = (unsigned short*)(ws + (size_t)NSITE * NSITE * 2);
    float* z0 = (float*)(ws + (size_t)NSITE * NSITE * 4);
    float* z1 = z0 + (size_t)NSAMP * NSITE;

    zero_out<<<NSAMP / NTHREADS, NTHREADS, 0, stream>>>(out);

    build_w<<<(unsigned)(((size_t)NSITE * NSITE) / NTHREADS), NTHREADS, 0, stream>>>(
        kern, Whi, Wlo);

    dim3 grid(NSITE / BN, NSAMP / BM);   // x = N-tiles (fast) for L2 reuse of A
    gemm_stage<0, 0><<<grid, NTHREADS, 0, stream>>>(x, nullptr, Whi, Wlo, z0, nullptr);
    gemm_stage<1, 0><<<grid, NTHREADS, 0, stream>>>(x, z0,      Whi, Wlo, z1, nullptr);
    gemm_stage<1, 1><<<grid, NTHREADS, 0, stream>>>(x, z1,      Whi, Wlo, nullptr, out);
}
